// LocalVisionTransformer_28681791603225
// MI455X (gfx1250) — matrix operand order repaired and checked by reading, not compiled, not run on hardware
//
#include <hip/hip_runtime.h>
#include <cmath>

// ---------------------------------------------------------------------------
// Local-window ViT forward for MI455X (gfx1250, wave32, WMMA bf16).
// B=32, N=197 (196 patches + cls), C=768, NH=12, hd=64, D=12, HID=3072.
// GEMMs: double-buffered LDS, b128 LDS traffic, packed bf16 converts,
// 4 back-to-back v_wmma per wave per K-step.
// ---------------------------------------------------------------------------

typedef __bf16 bf16;
typedef __attribute__((ext_vector_type(16))) __bf16 v16bf;
typedef __attribute__((ext_vector_type(8)))  __bf16 v8bf;
typedef __attribute__((ext_vector_type(8)))  float  v8f;

union V16U { v16bf v; v8bf h[2]; };

#define VIT_B   32
#define VIT_N   197
#define VIT_C   768
#define VIT_NH  12
#define VIT_HD  64
#define VIT_D   12
#define VIT_G   14
#define VIT_HID 3072
#define VIT_NC  1000

__device__ __forceinline__ v8bf cvt8(float4 a, float4 b) {
  v8bf r;
  r[0] = (bf16)a.x; r[1] = (bf16)a.y; r[2] = (bf16)a.z; r[3] = (bf16)a.w;
  r[4] = (bf16)b.x; r[5] = (bf16)b.y; r[6] = (bf16)b.z; r[7] = (bf16)b.w;
  return r;
}
__device__ __forceinline__ v8bf cvt8s(float4 a, float4 b, float s) {
  v8bf r;
  r[0] = (bf16)(a.x*s); r[1] = (bf16)(a.y*s); r[2] = (bf16)(a.z*s); r[3] = (bf16)(a.w*s);
  r[4] = (bf16)(b.x*s); r[5] = (bf16)(b.y*s); r[6] = (bf16)(b.z*s); r[7] = (bf16)(b.w*s);
  return r;
}

// ---------------- Generic fused GEMM: Out = f(A @ B + bias) (+res, +pos) ---
// 256 threads = 8 waves; 128x64 macro-tile; K in steps of 32 (bf16 WMMA).
#define GEMM_BM 128
#define GEMM_BN 64
#define GEMM_BK 32
#define A_LD 40      // halfs per A row: 32 + 8 pad -> 80B rows (16B multiple)
#define B_TILE 512   // halfs per fragment-native B tile (32 lanes x 16)

__global__ void __launch_bounds__(256)
gemm_wmma_kernel(const float* __restrict__ A, const float* __restrict__ Bm,
                 const float* __restrict__ bias, const float* __restrict__ res,
                 const float* __restrict__ pos, float* __restrict__ Out,
                 int M, int Nn, int K, int transB, int gelu, int rowMode)
{
  __shared__ __align__(16) bf16 As[2][GEMM_BM * A_LD];
  __shared__ __align__(16) bf16 BsW[2][4 * B_TILE];   // fragment-native B tiles

  const int tid  = threadIdx.x;
  const int wave = tid >> 5;
  const int lane = tid & 31;
  const int row0 = blockIdx.y * GEMM_BM;
  const int col0 = blockIdx.x * GEMM_BN;

  // A staging: thread -> (row, 16-k half). 256 thr cover 128 rows x 32 k.
  const int ar = tid >> 1, akg = tid & 1;
  const int agrow = row0 + ar;
  // B staging: thread -> (col, 8-k quarter). Writes fragment-native slots.
  const int bc = tid & 63, bkq = tid >> 6;             // bkq in 0..3
  const int gcol = col0 + bc;                          // Nn is mult of 64 here
  const int bTile = bc >> 4;
  const int bLane = (bc & 15) + ((bkq & 1) << 4);
  const int bSlot = (bTile * 32 + bLane) * 16 + ((bkq >> 1) << 3);

  const int nk = K >> 5;
  float4 fa0, fa1, fa2, fa3;   // staged A (16 floats)
  float4 gb0, gb1;             // staged B (8 floats)

  auto load_tile = [&](int k0) {
    const float4 z = {0.f, 0.f, 0.f, 0.f};
    fa0 = z; fa1 = z; fa2 = z; fa3 = z;
    if (agrow < M) {
      const float4* ap = reinterpret_cast<const float4*>(&A[(size_t)agrow * K + k0 + (akg << 4)]);
      fa0 = ap[0]; fa1 = ap[1]; fa2 = ap[2]; fa3 = ap[3];
    }
    if (transB) {
      const float4* bp = reinterpret_cast<const float4*>(&Bm[(size_t)gcol * K + k0 + (bkq << 3)]);
      gb0 = bp[0]; gb1 = bp[1];
    } else {
      const size_t kbase = (size_t)(k0 + (bkq << 3)) * Nn + gcol;
      gb0.x = Bm[kbase];            gb0.y = Bm[kbase + (size_t)Nn];
      gb0.z = Bm[kbase + 2u * Nn];  gb0.w = Bm[kbase + 3u * Nn];
      gb1.x = Bm[kbase + 4u * Nn];  gb1.y = Bm[kbase + 5u * Nn];
      gb1.z = Bm[kbase + 6u * Nn];  gb1.w = Bm[kbase + 7u * Nn];
    }
  };
  auto store_tile = [&](int p) {
    v8bf* ad = reinterpret_cast<v8bf*>(&As[p][ar * A_LD + (akg << 4)]);
    ad[0] = cvt8(fa0, fa1);
    ad[1] = cvt8(fa2, fa3);
    *reinterpret_cast<v8bf*>(&BsW[p][bSlot]) = cvt8(gb0, gb1);
  };

  v8f acc[4] = {};

  load_tile(0);
  store_tile(0);
  __syncthreads();

  for (int kt = 0; kt < nk; ++kt) {
    const int p = kt & 1;
    if (kt + 1 < nk) load_tile((kt + 1) << 5);   // overlap next-tile globals

    // Fragments: A + 4 B tiles into distinct registers (one clause, one wait),
    // then 4 independent WMMAs back-to-back.
    V16U ua;
    const bf16* ap = &As[p][(wave * 16 + (lane & 15)) * A_LD + ((lane >> 4) << 3)];
    ua.h[0] = *reinterpret_cast<const v8bf*>(ap);
    ua.h[1] = *reinterpret_cast<const v8bf*>(ap + 16);
    const bf16* bp = &BsW[p][lane * 16];
    V16U ub0, ub1, ub2, ub3;
    ub0.h[0] = *reinterpret_cast<const v8bf*>(bp);
    ub0.h[1] = *reinterpret_cast<const v8bf*>(bp + 8);
    ub1.h[0] = *reinterpret_cast<const v8bf*>(bp + B_TILE);
    ub1.h[1] = *reinterpret_cast<const v8bf*>(bp + B_TILE + 8);
    ub2.h[0] = *reinterpret_cast<const v8bf*>(bp + 2 * B_TILE);
    ub2.h[1] = *reinterpret_cast<const v8bf*>(bp + 2 * B_TILE + 8);
    ub3.h[0] = *reinterpret_cast<const v8bf*>(bp + 3 * B_TILE);
    ub3.h[1] = *reinterpret_cast<const v8bf*>(bp + 3 * B_TILE + 8);
    acc[0] = __builtin_amdgcn_wmma_f32_16x16x32_bf16(false, ua.v, false, ub0.v, (short)0, acc[0], false, false);
    acc[1] = __builtin_amdgcn_wmma_f32_16x16x32_bf16(false, ua.v, false, ub1.v, (short)0, acc[1], false, false);
    acc[2] = __builtin_amdgcn_wmma_f32_16x16x32_bf16(false, ua.v, false, ub2.v, (short)0, acc[2], false, false);
    acc[3] = __builtin_amdgcn_wmma_f32_16x16x32_bf16(false, ua.v, false, ub3.v, (short)0, acc[3], false, false);

    if (kt + 1 < nk) store_tile(p ^ 1);          // fill the other buffer
    __syncthreads();
  }

  // Epilogue: C/D tile layout -> row = (lane>>4)*8 + j, col = lane&15.
#pragma unroll
  for (int t = 0; t < 4; ++t) {
#pragma unroll
    for (int j = 0; j < 8; ++j) {
      const int row = row0 + wave * 16 + ((lane >> 4) << 3) + j;
      const int col = col0 + t * 16 + (lane & 15);
      if (row < M && col < Nn) {
        float v = acc[t][j];
        if (bias) v += bias[col];
        if (gelu) v = 0.5f * v * (1.f + erff(v * 0.70710678118f));
        const int orow = rowMode ? (row + row / 196 + 1) : row;
        if (pos) v += pos[(size_t)(orow % VIT_N) * Nn + col];
        if (res) v += res[(size_t)orow * Nn + col];
        Out[(size_t)orow * Nn + col] = v;
      }
    }
  }
}

// ---------------- LayerNorm: one block per row, C=768 ----------------------
__global__ void __launch_bounds__(256)
ln_kernel(const float* __restrict__ x, const float* __restrict__ w,
          const float* __restrict__ b, float* __restrict__ out, int xRowStride)
{
  const int row = blockIdx.x;
  const float* xr = x + (size_t)row * xRowStride;
  const int c0 = threadIdx.x;
  const float v0 = xr[c0], v1 = xr[c0 + 256], v2 = xr[c0 + 512];
  float s  = v0 + v1 + v2;
  float sq = v0 * v0 + v1 * v1 + v2 * v2;
#pragma unroll
  for (int off = 16; off > 0; off >>= 1) {
    s  += __shfl_down(s,  off, 32);
    sq += __shfl_down(sq, off, 32);
  }
  __shared__ float sb[8], qb[8];
  __shared__ float mu_s, rstd_s;
  if ((threadIdx.x & 31) == 0) { sb[threadIdx.x >> 5] = s; qb[threadIdx.x >> 5] = sq; }
  __syncthreads();
  if (threadIdx.x == 0) {
    float ts = 0.f, tq = 0.f;
#pragma unroll
    for (int i = 0; i < 8; ++i) { ts += sb[i]; tq += qb[i]; }
    const float mu  = ts * (1.f / VIT_C);
    const float var = tq * (1.f / VIT_C) - mu * mu;
    mu_s = mu; rstd_s = rsqrtf(var + 1e-5f);
  }
  __syncthreads();
  const float mu = mu_s, rstd = rstd_s;
  float* orow = out + (size_t)row * VIT_C;
  orow[c0]       = (v0 - mu) * rstd * w[c0]       + b[c0];
  orow[c0 + 256] = (v1 - mu) * rstd * w[c0 + 256] + b[c0 + 256];
  orow[c0 + 512] = (v2 - mu) * rstd * w[c0 + 512] + b[c0 + 512];
}

// ---------------- Local-window masked flash attention ----------------------
__device__ __forceinline__ bool attn_mask(int q, int k) {
  if (q >= VIT_N || k >= VIT_N) return false;
  if (q == 0 || k == 0) return true;
  const int qi = q - 1, ki = k - 1;
  const int qr = qi / VIT_G, qc = qi - qr * VIT_G;
  const int kr = ki / VIT_G, kc = ki - kr * VIT_G;
  int dr = qr - kr; if (dr < 0) dr = -dr;
  int dc = qc - kc; if (dc < 0) dc = -dc;
  return (dr <= 3) && (dc <= 3);
}

#define KS_LD 72    // 144B rows (16B multiple)
#define VT_LD 232   // 464B rows (16B multiple), 224 cols used + pad
#define P_LD  40    // 80B rows

__global__ void __launch_bounds__(128)
attn_kernel(const float* __restrict__ qkv, float* __restrict__ o)
{
  __shared__ __align__(16) bf16 Ks[VIT_N * KS_LD];    // K: 197 x 64, d contiguous
  __shared__ __align__(16) bf16 Vt[VIT_HD * VT_LD];   // V^T: 64 x 224 (zero tail)
  __shared__ __align__(16) bf16 Ps[4 * 16 * P_LD];    // per-wave 16x32 P tile

  const int b = blockIdx.x, h = blockIdx.y;
  const int tid = threadIdx.x, wave = tid >> 5, lane = tid & 31;

  // Stage K: (n, 16-d group) per thread -> float4 loads + 2 b128 stores.
  for (int g = tid; g < VIT_N * 4; g += 128) {
    const int n = g >> 2, dg = g & 3;
    const float4* kp4 = reinterpret_cast<const float4*>(
        qkv + ((((size_t)b * VIT_N + n) * 3 + 1) * VIT_NH + h) * VIT_HD + (dg << 4));
    v8bf* dst = reinterpret_cast<v8bf*>(&Ks[n * KS_LD + (dg << 4)]);
    dst[0] = cvt8(kp4[0], kp4[1]);
    dst[1] = cvt8(kp4[2], kp4[3]);
  }
  // Stage V transposed: (d, 16-n group) per thread -> 2 b128 stores.
  {
    const int d = tid & 63;
    for (int g = tid >> 6; g < 14; g += 2) {
      float4 g0, g1, g2, g3;
      float* bv0 = &g0.x;
#pragma unroll
      for (int i = 0; i < 16; ++i) {
        const int n = (g << 4) + i;
        bv0[i] = (n < VIT_N)
          ? qkv[((((size_t)b * VIT_N + n) * 3 + 2) * VIT_NH + h) * VIT_HD + d] : 0.f;
      }
      v8bf* dst = reinterpret_cast<v8bf*>(&Vt[d * VT_LD + (g << 4)]);
      dst[0] = cvt8(g0, g1);
      dst[1] = cvt8(g2, g3);
    }
  }
  __syncthreads();

  for (int qt = wave; qt < 13; qt += 4) {            // 13 query tiles of 16
    const int q0 = qt * 16;
    const int kb = (lane >> 4) << 4;
    int qrow = q0 + (lane & 15); if (qrow > VIT_N - 1) qrow = VIT_N - 1;
    const float* qp = qkv + (((size_t)b * VIT_N + qrow) * 3 * VIT_NH + h) * VIT_HD + kb;
    const float4* qp4 = reinterpret_cast<const float4*>(qp);
    V16U qa0, qa1;
    qa0.h[0] = cvt8s(qp4[0],  qp4[1],  0.125f);   // d = kb..kb+7
    qa0.h[1] = cvt8s(qp4[2],  qp4[3],  0.125f);   // d = kb+8..kb+15
    qa1.h[0] = cvt8s(qp4[8],  qp4[9],  0.125f);   // d = 32+kb..
    qa1.h[1] = cvt8s(qp4[10], qp4[11], 0.125f);   // d = 32+kb+8..

    float m[8], ls[8];
    v8f oacc[4] = {};
#pragma unroll
    for (int j = 0; j < 8; ++j) { m[j] = -3.0e38f; ls[j] = 0.f; }
    const int qrow_s = q0 + ((lane >> 4) << 3);    // S/O tile base row

    for (int kt = 0; kt < 7; ++kt) {               // 7 key tiles of 32
      const int k0 = kt * 32;
      v8f s0 = {}, s1 = {};
      {
        int kk0 = k0 + (lane & 15);      if (kk0 > VIT_N - 1) kk0 = VIT_N - 1;
        int kk1 = k0 + 16 + (lane & 15); if (kk1 > VIT_N - 1) kk1 = VIT_N - 1;
        const bf16* kpA = &Ks[kk0 * KS_LD + ((lane >> 4) << 4)];
        const bf16* kpB = &Ks[kk1 * KS_LD + ((lane >> 4) << 4)];
        V16U ba0, ba1, bb0, bb1;
        ba0.h[0] = *reinterpret_cast<const v8bf*>(kpA);
        ba0.h[1] = *reinterpret_cast<const v8bf*>(kpA + 8);
        ba1.h[0] = *reinterpret_cast<const v8bf*>(kpA + 32);
        ba1.h[1] = *reinterpret_cast<const v8bf*>(kpA + 40);
        bb0.h[0] = *reinterpret_cast<const v8bf*>(kpB);
        bb0.h[1] = *reinterpret_cast<const v8bf*>(kpB + 8);
        bb1.h[0] = *reinterpret_cast<const v8bf*>(kpB + 32);
        bb1.h[1] = *reinterpret_cast<const v8bf*>(kpB + 40);
        s0 = __builtin_amdgcn_wmma_f32_16x16x32_bf16(false, qa0.v, false, ba0.v, (short)0, s0, false, false);
        s1 = __builtin_amdgcn_wmma_f32_16x16x32_bf16(false, qa0.v, false, bb0.v, (short)0, s1, false, false);
        s0 = __builtin_amdgcn_wmma_f32_16x16x32_bf16(false, qa1.v, false, ba1.v, (short)0, s0, false, false);
        s1 = __builtin_amdgcn_wmma_f32_16x16x32_bf16(false, qa1.v, false, bb1.v, (short)0, s1, false, false);
      }

      // Masked online softmax; rows live across 16 lanes of one half-wave.
      bf16* Pw = &Ps[wave * 16 * P_LD];
#pragma unroll
      for (int j = 0; j < 8; ++j) {
        const int qg  = qrow_s + j;
        const int kg0 = k0 + (lane & 15);
        const float a0 = attn_mask(qg, kg0)      ? s0[j] : -3.0e38f;
        const float a1 = attn_mask(qg, kg0 + 16) ? s1[j] : -3.0e38f;
        float v = fmaxf(a0, a1);
#pragma unroll
        for (int off = 1; off < 16; off <<= 1) v = fmaxf(v, __shfl_xor(v, off, 32));
        const float mn   = fmaxf(m[j], v);
        const float corr = __expf(m[j] - mn);
        const float p0 = __expf(a0 - mn);
        const float p1 = __expf(a1 - mn);
        float rs = p0 + p1;
#pragma unroll
        for (int off = 1; off < 16; off <<= 1) rs += __shfl_xor(rs, off, 32);
        ls[j] = ls[j] * corr + rs;
        m[j]  = mn;
#pragma unroll
        for (int t = 0; t < 4; ++t) oacc[t][j] *= corr;
        const int prow = ((lane >> 4) << 3) + j;
        Pw[prow * P_LD + (lane & 15)]      = (bf16)p0;
        Pw[prow * P_LD + 16 + (lane & 15)] = (bf16)p1;
      }
      asm volatile("s_wait_dscnt 0x0" ::: "memory");  // P LDS RAW (same wave)

      // O += P @ V : A-frag from Ps, 4 V-frags into distinct regs, 4 WMMAs.
      V16U pa;
      const bf16* pp = &Ps[(wave * 16 + (lane & 15)) * P_LD + ((lane >> 4) << 4)];
      pa.h[0] = *reinterpret_cast<const v8bf*>(pp);
      pa.h[1] = *reinterpret_cast<const v8bf*>(pp + 8);
      const bf16* vp = &Vt[(lane & 15) * VT_LD + k0 + ((lane >> 4) << 4)];
      V16U uv0, uv1, uv2, uv3;
      uv0.h[0] = *reinterpret_cast<const v8bf*>(vp);
      uv0.h[1] = *reinterpret_cast<const v8bf*>(vp + 8);
      uv1.h[0] = *reinterpret_cast<const v8bf*>(vp + 16 * VT_LD);
      uv1.h[1] = *reinterpret_cast<const v8bf*>(vp + 16 * VT_LD + 8);
      uv2.h[0] = *reinterpret_cast<const v8bf*>(vp + 32 * VT_LD);
      uv2.h[1] = *reinterpret_cast<const v8bf*>(vp + 32 * VT_LD + 8);
      uv3.h[0] = *reinterpret_cast<const v8bf*>(vp + 48 * VT_LD);
      uv3.h[1] = *reinterpret_cast<const v8bf*>(vp + 48 * VT_LD + 8);
      oacc[0] = __builtin_amdgcn_wmma_f32_16x16x32_bf16(false, pa.v, false, uv0.v, (short)0, oacc[0], false, false);
      oacc[1] = __builtin_amdgcn_wmma_f32_16x16x32_bf16(false, pa.v, false, uv1.v, (short)0, oacc[1], false, false);
      oacc[2] = __builtin_amdgcn_wmma_f32_16x16x32_bf16(false, pa.v, false, uv2.v, (short)0, oacc[2], false, false);
      oacc[3] = __builtin_amdgcn_wmma_f32_16x16x32_bf16(false, pa.v, false, uv3.v, (short)0, oacc[3], false, false);
    }

    // Normalize and store O -> (B,N,C) with col = h*64 + d.
#pragma unroll
    for (int j = 0; j < 8; ++j) {
      const int qg = qrow_s + j;
      if (qg < VIT_N) {
        const float inv = 1.f / ls[j];
#pragma unroll
        for (int t = 0; t < 4; ++t) {
          o[((size_t)b * VIT_N + qg) * VIT_C + h * VIT_HD + t * 16 + (lane & 15)] =
              oacc[t][j] * inv;
        }
      }
    }
  }
}

// ---------------- Small helper kernels -------------------------------------
__global__ void im2col_kernel(const float* __restrict__ x, float* __restrict__ xp)
{
  const int idx = blockIdx.x * blockDim.x + threadIdx.x;
  if (idx >= VIT_B * 196 * VIT_C) return;
  const int b = idx / (196 * VIT_C);
  const int rem = idx - b * 196 * VIT_C;
  const int n = rem / VIT_C;
  const int k = rem - n * VIT_C;
  const int ch = k >> 8, r2 = k & 255, py = r2 >> 4, px = r2 & 15;
  const int gr = n / VIT_G, gc = n - gr * VIT_G;
  xp[idx] = x[(((size_t)b * 3 + ch) * 224 + (gr * 16 + py)) * 224 + (gc * 16 + px)];
}

__global__ void cls_pos_kernel(const float* __restrict__ cls,
                               const float* __restrict__ pos, float* __restrict__ t)
{
  const int idx = blockIdx.x * blockDim.x + threadIdx.x;
  if (idx >= VIT_B * VIT_C) return;
  const int b = idx / VIT_C, c = idx - b * VIT_C;
  t[(size_t)b * VIT_N * VIT_C + c] = cls[c] + pos[c];
}

__global__ void head_kernel(const float* __restrict__ tcls, const float* __restrict__ W,
                            const float* __restrict__ bias, float* __restrict__ out)
{
  const int idx = blockIdx.x * blockDim.x + threadIdx.x;
  if (idx >= VIT_B * VIT_NC) return;
  const int b = idx / VIT_NC, c = idx - b * VIT_NC;
  float s = bias[c];
  const float* tr = tcls + (size_t)b * VIT_C;
  for (int k = 0; k < VIT_C; ++k) s += tr[k] * W[(size_t)k * VIT_NC + c];
  out[idx] = s;
}

// ---------------- Launcher --------------------------------------------------
extern "C" void kernel_launch(void* const* d_in, const int* in_sizes, int n_in,
                              void* d_out, int out_size, void* d_ws, size_t ws_size,
                              hipStream_t stream)
{
  (void)in_sizes; (void)n_in; (void)out_size; (void)ws_size;
  const float* x         = (const float*)d_in[0];
  const float* patch_w   = (const float*)d_in[1];
  const float* patch_b   = (const float*)d_in[2];
  const float* cls_token = (const float*)d_in[3];
  const float* pos_embed = (const float*)d_in[4];
  const float* ln1_w     = (const float*)d_in[5];
  const float* ln1_b     = (const float*)d_in[6];
  const float* qkv_w     = (const float*)d_in[7];
  const float* proj_w    = (const float*)d_in[8];
  const float* proj_b    = (const float*)d_in[9];
  const float* ln2_w     = (const float*)d_in[10];
  const float* ln2_b     = (const float*)d_in[11];
  const float* mlp_w1    = (const float*)d_in[12];
  const float* mlp_b1    = (const float*)d_in[13];
  const float* mlp_w2    = (const float*)d_in[14];
  const float* mlp_b2    = (const float*)d_in[15];
  const float* norm_w    = (const float*)d_in[16];
  const float* norm_b    = (const float*)d_in[17];
  const float* head_w    = (const float*)d_in[18];
  const float* head_b    = (const float*)d_in[19];
  float* out = (float*)d_out;

  const size_t M  = (size_t)VIT_B * VIT_N;          // 6304
  float* ws  = (float*)d_ws;
  float* xp  = ws;                                   // 6272 x 768
  float* t   = xp  + (size_t)VIT_B * 196 * VIT_C;    // 6304 x 768
  float* h   = t   + M * VIT_C;                      // 6304 x 768
  float* qkv = h   + M * VIT_C;                      // 6304 x 2304
  float* ob  = qkv + M * 3 * VIT_C;                  // 6304 x 768
  float* hid = ob  + M * VIT_C;                      // 6304 x 3072
  float* tc  = hid + M * VIT_HID;                    // 32 x 768

  // Patch embedding: im2col, cls+pos row, then GEMM (W transposed) fused
  // with patch bias + pos_embed add and 196->197 row remap.
  im2col_kernel<<<(VIT_B * 196 * VIT_C + 255) / 256, 256, 0, stream>>>(x, xp);
  cls_pos_kernel<<<(VIT_B * VIT_C + 255) / 256, 256, 0, stream>>>(cls_token, pos_embed, t);
  gemm_wmma_kernel<<<dim3(12, 49), 256, 0, stream>>>(
      xp, patch_w, patch_b, nullptr, pos_embed, t,
      VIT_B * 196, VIT_C, VIT_C, /*transB=*/1, /*gelu=*/0, /*rowMode=*/1);

  for (int i = 0; i < VIT_D; ++i) {
    ln_kernel<<<(int)M, 256, 0, stream>>>(t, ln1_w + i * VIT_C, ln1_b + i * VIT_C, h, VIT_C);
    gemm_wmma_kernel<<<dim3(36, 50), 256, 0, stream>>>(
        h, qkv_w + (size_t)i * VIT_C * 3 * VIT_C, nullptr, nullptr, nullptr, qkv,
        (int)M, 3 * VIT_C, VIT_C, 0, 0, 0);
    attn_kernel<<<dim3(VIT_B, VIT_NH), 128, 0, stream>>>(qkv, ob);
    gemm_wmma_kernel<<<dim3(12, 50), 256, 0, stream>>>(
        ob, proj_w + (size_t)i * VIT_C * VIT_C, proj_b + i * VIT_C, t, nullptr, t,
        (int)M, VIT_C, VIT_C, 0, 0, 0);
    ln_kernel<<<(int)M, 256, 0, stream>>>(t, ln2_w + i * VIT_C, ln2_b + i * VIT_C, h, VIT_C);
    gemm_wmma_kernel<<<dim3(48, 50), 256, 0, stream>>>(
        h, mlp_w1 + (size_t)i * VIT_C * VIT_HID, mlp_b1 + i * VIT_HID, nullptr, nullptr, hid,
        (int)M, VIT_HID, VIT_C, 0, /*gelu=*/1, 0);
    gemm_wmma_kernel<<<dim3(12, 50), 256, 0, stream>>>(
        hid, mlp_w2 + (size_t)i * VIT_HID * VIT_C, mlp_b2 + i * VIT_C, t, nullptr, t,
        (int)M, VIT_C, VIT_HID, 0, 0, 0);
  }

  // Final LN on cls rows only (LN is row-wise), then classifier head.
  ln_kernel<<<VIT_B, 256, 0, stream>>>(t, norm_w, norm_b, tc, VIT_N * VIT_C);
  head_kernel<<<(VIT_B * VIT_NC + 255) / 256, 256, 0, stream>>>(tc, head_w, head_b, out);
}
